// GrapherModule_44787918962777
// MI455X (gfx1250) — matrix operand order repaired and checked by reading, not compiled, not run on hardware
//
#include <hip/hip_runtime.h>
#include <cfloat>
#include <climits>

// ---------------- CDNA5 (gfx1250) wave32 WMMA types ----------------
typedef __attribute__((ext_vector_type(16))) __bf16 v16bf;
typedef __attribute__((ext_vector_type(8)))  __bf16 v8bf;
typedef __attribute__((ext_vector_type(8)))  float  v8f;

#define BB     8
#define CC     96
#define NN     3136      // 56*56
#define HCC    192
#define KNN_K  9
#define NTILES 196       // NN/16

// =====================================================================
// Kernel 1: fc1 (1x1 conv, 96->96) + BN1 -> f[b][n][c] (f32 + bf16 copy)
// and sq[b][n] = sum_c f^2. One block (96 threads) per (b,n).
// =====================================================================
__global__ void k_fc1(const float* __restrict__ x, const float* __restrict__ w1,
                      const float* __restrict__ b1,
                      const float* __restrict__ g,  const float* __restrict__ be,
                      const float* __restrict__ mm, const float* __restrict__ vv,
                      float* __restrict__ f, __bf16* __restrict__ fbf,
                      float* __restrict__ sq)
{
    __shared__ float xcol[CC];
    __shared__ float red[CC];
    int bn = blockIdx.x;
    int b = bn / NN, n = bn % NN;
    int o = threadIdx.x;                       // 0..95
    xcol[o] = x[((size_t)b * CC + o) * NN + n];
    __syncthreads();
    const float* wr = w1 + (size_t)o * CC;
    float acc = b1[o];
#pragma unroll 8
    for (int c = 0; c < CC; ++c) acc += wr[c] * xcol[c];
    float inv = g[o] / sqrtf(vv[o] + 1e-5f);
    float val = acc * inv + (be[o] - mm[o] * inv);
    size_t fo = ((size_t)b * NN + n) * CC + o;
    f[fo]   = val;
    fbf[fo] = (__bf16)val;
    red[o]  = val * val;
    __syncthreads();
    if (o < 32) {                               // wave32 reduction
        float s = red[o] + red[o + 32] + red[o + 64];
#pragma unroll
        for (int d = 16; d > 0; d >>= 1) s += __shfl_xor(s, d, 32);
        if (o == 0) sq[(size_t)b * NN + n] = s;
    }
}

// =====================================================================
// Kernel: f32 -> bf16 conversion for w_g (B-operand: row-major [oc][ic],
// contiguous ic runs match WMMA B lane layout).
// =====================================================================
__global__ void k_cvt(const float* __restrict__ w, __bf16* __restrict__ wb, int nelem)
{
    int i = blockIdx.x * blockDim.x + threadIdx.x;
    if (i < nelem) wb[i] = (__bf16)w[i];
}

// =====================================================================
// Kernel 2: pairwise distances via WMMA + fused top-9.
// One workgroup (8 waves) per 16-row tile of one batch. The full
// 16 x 3136 f32 distance stripe lives in LDS (200 KB) -- only possible
// on CDNA5's 320 KB-per-WGP LDS. dist = sq_n + sq_m - 2 * <f_n, f_m>,
// dot computed with v_wmma_f32_16x16x32_bf16 (K=96 -> 3 chained WMMAs),
// A operand cached in VGPRs across all 196 column tiles.
// =====================================================================
__global__ __launch_bounds__(256)
void k_dist_topk(const __bf16* __restrict__ fbf, const float* __restrict__ sq,
                 int* __restrict__ knn)
{
    extern __shared__ float smem[];            // [16] sqrow + [16*NN] dist
    float* sqrow = smem;
    float* dist  = smem + 16;
    int blk = blockIdx.x;
    int b  = blk / NTILES;
    int n0 = (blk % NTILES) * 16;
    int tid  = threadIdx.x;
    int wave = tid >> 5;
    int lane = tid & 31;
    int lm   = lane & 15;
    int hi   = lane >> 4;

    if (tid < 16) sqrow[tid] = sq[(size_t)b * NN + n0 + tid];
    __syncthreads();

    // A tile (rows n0..n0+15) cached in registers: 3 K-chunks of 32.
    // A layout: lane holds M = lane%16; K run {0..15} (lo half-wave)
    // or {16..31} (hi half-wave).
    const __bf16* arow = fbf + ((size_t)b * NN + n0 + lm) * CC;
    v16bf A[3];
#pragma unroll
    for (int kk = 0; kk < 3; ++kk) {
        v8bf lo = *reinterpret_cast<const v8bf*>(arow + kk * 32 + 16 * hi);
        v8bf hh = *reinterpret_cast<const v8bf*>(arow + kk * 32 + 16 * hi + 8);
#pragma unroll
        for (int e = 0; e < 8; ++e) { A[kk][e] = lo[e]; A[kk][e + 8] = hh[e]; }
    }

    for (int t = wave; t < NTILES; t += 8) {
        int m0 = t * 16;
        // B layout: lane holds N = lane%16, contiguous K run of 16 chosen by lane/16.
        const __bf16* brow = fbf + ((size_t)b * NN + m0 + lm) * CC;
        v8f acc = {};
#pragma unroll
        for (int kk = 0; kk < 3; ++kk) {
            v16bf Bm = *reinterpret_cast<const v16bf*>(brow + kk * 32 + 16 * hi);
            acc = __builtin_amdgcn_wmma_f32_16x16x32_bf16(
                false, A[kk], false, Bm, (short)0, acc, false, false);
        }
        float sqc = sq[(size_t)b * NN + m0 + lm];
#pragma unroll
        for (int i = 0; i < 8; ++i) {          // C layout: VGPR i -> rows i / i+8
            int r = i + 8 * hi;
            dist[r * NN + m0 + lm] = sqrow[r] + sqc - 2.0f * acc[i];
        }
    }
    __syncthreads();

    // Top-9 (smallest dist, ties -> lower index, matching top_k(-dist)).
    // Each wave owns 2 rows; per-lane sorted insertion list over a strided
    // 1/32 slice, then 9 rounds of wave32 shfl_xor min-merge.
    for (int rr = 0; rr < 2; ++rr) {
        int r = 2 * wave + rr;
        float tv[KNN_K]; int ti[KNN_K];
#pragma unroll
        for (int q = 0; q < KNN_K; ++q) { tv[q] = FLT_MAX; ti[q] = INT_MAX; }
        const float* drow = dist + (size_t)r * NN;
        for (int s = 0; s < NN / 32; ++s) {
            int m = lane + 32 * s;
            float d = drow[m];
            if (d < tv[KNN_K - 1] || (d == tv[KNN_K - 1] && m < ti[KNN_K - 1])) {
                tv[KNN_K - 1] = d; ti[KNN_K - 1] = m;
#pragma unroll
                for (int p = KNN_K - 1; p > 0; --p) {
                    bool sw = (tv[p] < tv[p - 1]) ||
                              (tv[p] == tv[p - 1] && ti[p] < ti[p - 1]);
                    if (sw) {
                        float tf = tv[p]; tv[p] = tv[p - 1]; tv[p - 1] = tf;
                        int   tt = ti[p]; ti[p] = ti[p - 1]; ti[p - 1] = tt;
                    }
                }
            }
        }
        int* orow = knn + ((size_t)b * NN + n0 + r) * KNN_K;
        for (int r9 = 0; r9 < KNN_K; ++r9) {
            float bv = tv[0]; int bi = ti[0];
#pragma unroll
            for (int off = 16; off > 0; off >>= 1) {
                float ov = __shfl_xor(bv, off, 32);
                int   oi = __shfl_xor(bi, off, 32);
                if (ov < bv || (ov == bv && oi < bi)) { bv = ov; bi = oi; }
            }
            if (lane == 0) orow[r9] = bi;
            if (tv[0] == bv && ti[0] == bi) {   // winner pops its head
#pragma unroll
                for (int p = 0; p < KNN_K - 1; ++p) { tv[p] = tv[p + 1]; ti[p] = ti[p + 1]; }
                tv[KNN_K - 1] = FLT_MAX; ti[KNN_K - 1] = INT_MAX;
            }
        }
    }
}

// =====================================================================
// Kernel 3: edge conv. One workgroup per 16 pixels -> 144 edge rows
// (exactly 9 WMMA M-tiles). Edge matrix [x_i | x_j - x_i] built in LDS
// as bf16, GEMM vs w_g^T (192x192) with WMMA, bias, max over k=9,
// BN_g + exact-erf GELU -> eg[b][n][oc].
// =====================================================================
__global__ __launch_bounds__(256)
void k_edge(const float* __restrict__ f, const int* __restrict__ knn,
            const __bf16* __restrict__ wgb, const float* __restrict__ bg,
            const float* __restrict__ gg, const float* __restrict__ gb,
            const float* __restrict__ gm, const float* __restrict__ gv,
            float* __restrict__ eg)
{
    extern __shared__ char smemc[];
    __bf16* E  = reinterpret_cast<__bf16*>(smemc);                  // 144*192 bf16
    float*  E2 = reinterpret_cast<float*>(smemc + 144 * HCC * 2);   // 144*192 f32
    int blk = blockIdx.x;
    int b  = blk / NTILES;
    int n0 = (blk % NTILES) * 16;
    int tid = threadIdx.x;

    // Phase 1: build edge features (row r = 9*j + k).
    for (int t = tid; t < 144 * HCC; t += 256) {
        int r = t / HCC, c = t % HCC;
        int j = r / KNN_K, k = r % KNN_K;
        int n = n0 + j;
        float v;
        if (c < CC) {
            v = f[((size_t)b * NN + n) * CC + c];
        } else {
            int cj = c - CC;
            int nj = knn[((size_t)b * NN + n) * KNN_K + k];
            v = f[((size_t)b * NN + nj) * CC + cj] - f[((size_t)b * NN + n) * CC + cj];
        }
        E[t] = (__bf16)v;
    }
    __syncthreads();

    // Phase 2: 9 x 12 output tiles, K = 192 -> 6 chained WMMAs each.
    int wave = tid >> 5, lane = tid & 31, lm = lane & 15, hi = lane >> 4;
    for (int T = wave; T < 9 * 12; T += 8) {
        int mt = T / 12, nt = T % 12;
        int col = nt * 16 + lm;
        v8f acc = {};
#pragma unroll
        for (int kk = 0; kk < 6; ++kk) {
            const __bf16* ar = E + (mt * 16 + lm) * HCC;
            v8bf lo = *reinterpret_cast<const v8bf*>(ar + kk * 32 + 16 * hi);
            v8bf hh = *reinterpret_cast<const v8bf*>(ar + kk * 32 + 16 * hi + 8);
            v16bf Am;
#pragma unroll
            for (int e = 0; e < 8; ++e) { Am[e] = lo[e]; Am[e + 8] = hh[e]; }
            v16bf Bm = *reinterpret_cast<const v16bf*>(
                wgb + (size_t)col * HCC + kk * 32 + 16 * hi);
            acc = __builtin_amdgcn_wmma_f32_16x16x32_bf16(
                false, Am, false, Bm, (short)0, acc, false, false);
        }
        float bias = bg[col];
#pragma unroll
        for (int i = 0; i < 8; ++i) {
            int row = mt * 16 + i + 8 * hi;
            E2[row * HCC + col] = acc[i] + bias;
        }
    }
    __syncthreads();

    // Phase 3: max over k, BN_g, exact GELU.
    for (int t = tid; t < 16 * HCC; t += 256) {
        int j = t / HCC, oc = t % HCC;
        float mx = -FLT_MAX;
#pragma unroll
        for (int k = 0; k < KNN_K; ++k) mx = fmaxf(mx, E2[(j * KNN_K + k) * HCC + oc]);
        float inv = gg[oc] / sqrtf(gv[oc] + 1e-5f);
        float val = mx * inv + (gb[oc] - gm[oc] * inv);
        float gel = 0.5f * val * (1.0f + erff(val * 0.70710678118654752f));
        eg[((size_t)b * NN + n0 + j) * HCC + oc] = gel;
    }
}

// =====================================================================
// Kernel 4: fc2 (192->96) + BN2 + residual, output in (B,C,H,W) layout.
// =====================================================================
__global__ void k_fc2(const float* __restrict__ eg, const float* __restrict__ w2,
                      const float* __restrict__ b2,
                      const float* __restrict__ g,  const float* __restrict__ be,
                      const float* __restrict__ mm, const float* __restrict__ vv,
                      const float* __restrict__ x, float* __restrict__ out)
{
    __shared__ float erow[HCC];
    int bn = blockIdx.x;
    int b = bn / NN, n = bn % NN;
    int o = threadIdx.x;                       // 0..95
    erow[o]      = eg[((size_t)b * NN + n) * HCC + o];
    erow[o + CC] = eg[((size_t)b * NN + n) * HCC + o + CC];
    __syncthreads();
    const float* wr = w2 + (size_t)o * HCC;
    float acc = b2[o];
#pragma unroll 8
    for (int c = 0; c < HCC; ++c) acc += wr[c] * erow[c];
    float inv = g[o] / sqrtf(vv[o] + 1e-5f);
    float val = acc * inv + (be[o] - mm[o] * inv);
    size_t xo = ((size_t)b * CC + o) * NN + n;
    out[xo] = val + x[xo];
}

// =====================================================================
extern "C" void kernel_launch(void* const* d_in, const int* in_sizes, int n_in,
                              void* d_out, int out_size, void* d_ws, size_t ws_size,
                              hipStream_t stream)
{
    const float* x    = (const float*)d_in[0];
    const float* w1   = (const float*)d_in[1];
    const float* b1   = (const float*)d_in[2];
    const float* bn1g = (const float*)d_in[3];
    const float* bn1b = (const float*)d_in[4];
    const float* bn1m = (const float*)d_in[5];
    const float* bn1v = (const float*)d_in[6];
    const float* wg   = (const float*)d_in[7];
    const float* bg   = (const float*)d_in[8];
    const float* bngg = (const float*)d_in[9];
    const float* bngb = (const float*)d_in[10];
    const float* bngm = (const float*)d_in[11];
    const float* bngv = (const float*)d_in[12];
    const float* w2   = (const float*)d_in[13];
    const float* b2   = (const float*)d_in[14];
    const float* bn2g = (const float*)d_in[15];
    const float* bn2b = (const float*)d_in[16];
    const float* bn2m = (const float*)d_in[17];
    const float* bn2v = (const float*)d_in[18];
    float* out = (float*)d_out;
    (void)in_sizes; (void)n_in; (void)out_size; (void)ws_size;

    // Workspace carving (~35 MB total), 256-byte aligned regions so the
    // bf16 vector loads (v8bf/v16bf) stay 16/32-byte aligned.
    char* ws = (char*)d_ws;
    size_t off = 0;
    auto carve = [&](size_t bytes) -> char* {
        char* p = ws + off;
        off = (off + bytes + 255) & ~(size_t)255;
        return p;
    };
    float*  f   = (float*) carve((size_t)BB * NN * CC  * 4);
    __bf16* fb  = (__bf16*)carve((size_t)BB * NN * CC  * 2);
    float*  sq  = (float*) carve((size_t)BB * NN       * 4);
    int*    knn = (int*)   carve((size_t)BB * NN * KNN_K * 4);
    float*  eg  = (float*) carve((size_t)BB * NN * HCC * 4);
    __bf16* wgb = (__bf16*)carve((size_t)HCC * HCC     * 2);

    k_fc1<<<dim3(BB * NN), dim3(CC), 0, stream>>>(
        x, w1, b1, bn1g, bn1b, bn1m, bn1v, f, fb, sq);

    k_cvt<<<dim3((HCC * HCC + 255) / 256), dim3(256), 0, stream>>>(wg, wgb, HCC * HCC);

    // 16 + 16*3136 floats = 200768 B of dynamic LDS (CDNA5: up to 320 KB/WGP).
    size_t sh2 = (16 + 16 * (size_t)NN) * sizeof(float);
    k_dist_topk<<<dim3(BB * NTILES), dim3(256), sh2, stream>>>(fb, sq, knn);

    // 144*192 bf16 + 144*192 f32 = 165888 B of dynamic LDS.
    size_t sh3 = (size_t)144 * HCC * 2 + (size_t)144 * HCC * 4;
    k_edge<<<dim3(BB * NTILES), dim3(256), sh3, stream>>>(
        f, knn, wgb, bg, bngg, bngb, bngm, bngv, eg);

    k_fc2<<<dim3(BB * NN), dim3(CC), 0, stream>>>(
        eg, w2, b2, bn2g, bn2b, bn2m, bn2v, x, out);
}
